// Conv4d_31190052504125
// MI455X (gfx1250) — matrix operand order repaired and checked by reading, not compiled, not run on hardware
//
#include <hip/hip_runtime.h>
#include <hip/hip_bf16.h>
#include <stdint.h>

typedef __attribute__((ext_vector_type(16))) _Float16 v16h;
typedef __attribute__((ext_vector_type(8)))  float    v8f;

// Problem constants
#define BN   2
#define CI   8
#define TT   16
#define DD   40
#define HH   40
#define WW   40
#define CO   32
#define OO   16
#define KK   3

// Tiling
#define HB    4                 // h rows per workgroup
#define PH    (HB + 2)          // patch h extent (halo)
#define PW    (WW + 2)          // patch w extent (halo)
#define PATCH (KK * KK * PH * PW * CI)   // 3*3*6*42*8 = 18144 halves
#define NWAVE 10                // 10 N-tiles of 16 = 4*40 positions
#define BLOCK (NWAVE * 32)

#define RPAD  672               // padded reduction length (21 chunks of 32)
#define NCHUNK 21
#define ASTRIDE (RPAD * 2)      // bytes per co row in f16 weight table (1344)
#define COSTRIDE ((size_t)OO * DD * HH * WW)   // out stride per co

// ---------------------------------------------------------------------------
// Kernel 1: convert weights fp32 -> f16 table  wf16[co][r], r = tap*8 + ci,
// tap = ((k*3+kd)*3+kh)*3+kw.  Rows r >= 648 are zero (K padding).
// ---------------------------------------------------------------------------
__global__ void wcvt_kernel(const float* __restrict__ wsrc,
                            _Float16* __restrict__ wdst) {
    int i = blockIdx.x * 256 + threadIdx.x;
    if (i >= CO * RPAD) return;
    int co = i / RPAD;
    int r  = i - co * RPAD;
    float v = 0.0f;
    if (r < 648) {
        int tap = r >> 3;
        int ci  = r & 7;
        int k   = tap / 27;
        int rem = tap - k * 27;          // kd*9 + kh*3 + kw
        v = wsrc[(size_t)((k * CO + co) * CI + ci) * 27 + rem];
    }
    wdst[i] = (_Float16)v;
}

// patch byte-offset contributed by a tap, layout [t][dz][hh][ww][ci] (f16)
__device__ constexpr int tap_off_b(int tap) {
    int k   = tap / 27;
    int rem = tap - k * 27;
    int kd  = rem / 9;
    int r9  = rem - kd * 9;
    int kh  = r9 / 3;
    int kw  = r9 - kh * 3;
    return ((((k * 3 + kd) * PH + kh) * PW + kw) * CI) * 2;
}

// ---------------------------------------------------------------------------
// Kernel 2: main implicit-GEMM conv.  One workgroup = (b, o, d, h-block).
// Each of the 10 waves computes a 32(co) x 16(pos) output tile via WMMA.
// ---------------------------------------------------------------------------
__global__ __launch_bounds__(BLOCK)
void conv4d_kernel(const float* __restrict__ x,
                   const _Float16* __restrict__ wf16,
                   const float* __restrict__ bias,
                   float* __restrict__ out) {
    __shared__ _Float16 plds[PATCH];
    __shared__ float    bsum[CO];

    int bid = blockIdx.x;
    int hb = bid % (HH / HB);  bid /= (HH / HB);
    int d  = bid % DD;         bid /= DD;
    int o  = bid % OO;         bid /= OO;
    int b  = bid;
    int h0 = hb * HB;

    const int tid = threadIdx.x;

    // ---- stage x patch (with zero halo) into LDS as f16, ci innermost ----
    for (int s = tid; s < PATCH; s += BLOCK) {
        int ww = s % PW;  int q = s / PW;      // ww fastest -> coalesced x reads
        int ci = q % CI;  q /= CI;
        int hh = q % PH;  q /= PH;
        int dz = q % KK;  q /= KK;
        int t  = q;                            // 0..2
        int tt = o - 1 + t;
        int dd = d - 1 + dz;
        int hg = h0 - 1 + hh;
        int wg = ww - 1;
        float v = 0.0f;
        if ((unsigned)tt < (unsigned)TT && (unsigned)dd < (unsigned)DD &&
            (unsigned)hg < (unsigned)HH && (unsigned)wg < (unsigned)WW) {
            size_t xi = ((((size_t)b * CI + ci) * TT + tt) * DD + dd) * (HH * WW)
                        + (size_t)hg * WW + wg;
            v = x[xi];
        }
        int li = (((t * KK + dz) * PH + hh) * PW + ww) * CI + ci;
        plds[li] = (_Float16)v;
    }

    // ---- per-co bias sum over valid temporal taps ----
    if (tid < CO) {
        float s = 0.0f;
        #pragma unroll
        for (int k = 0; k < KK; ++k) {
            int tt = o - 1 + k;
            if (tt >= 0 && tt < TT) s += bias[k * CO + tid];
        }
        bsum[tid] = s;
    }
    __syncthreads();

    // ---- per-wave GEMM tile ----
    const int wave = tid >> 5;
    const int lane = tid & 31;
    const bool isHi = lane >= 16;
    const int nloc = lane & 15;
    const int p    = wave * 16 + nloc;      // 0..159
    const int h_off = p / WW;               // 0..3
    const int w     = p - h_off * WW;       // 0..39
    const int m     = lane & 15;            // A-matrix row (co within tile)

    // per-lane base pointers (all chunk-dependent offsets become immediates)
    const char* bptr  = (const char*)plds + ((h_off * PW + w) * CI) * 2;
    const char* aptr0 = (const char*)wf16 + m * ASTRIDE + (isHi ? 16 : 0);
    const char* aptr1 = aptr0 + 16 * ASTRIDE;

    v8f acc0 = {};
    v8f acc1 = {};

    #pragma unroll
    for (int c = 0; c < NCHUNK; ++c) {
        // ---------------- B fragment (32K x 16N, f16) ----------------
        // lanes 0-15 hold chunk rows 0..7,16..23 (taps 4c,4c+2); lanes 16-31 rows 8..15,24..31 (taps 4c+1,4c+3)
        const int tapLoA = 4 * c,     tapLoB = 4 * c + 1;   // compile-time
        const int tapHiA = 4 * c + 2, tapHiB = 4 * c + 3;
        const bool vLoA = tapLoA < 81, vLoB = tapLoB < 81;  // fold per iteration
        const bool vHiA = tapHiA < 81, vHiB = tapHiB < 81;
        const int offLoA = tap_off_b(vLoA ? tapLoA : 0);
        const int offLoB = tap_off_b(vLoB ? tapLoB : 0);
        const int offHiA = tap_off_b(vHiA ? tapHiA : 0);
        const int offHiB = tap_off_b(vHiB ? tapHiB : 0);

        const int  offA = isHi ? offLoB : offLoA;
        const int  offB = isHi ? offHiB : offHiA;
        const bool vA   = isHi ? vLoB : vLoA;
        const bool vB   = isHi ? vHiB : vHiA;

        union { v16h h; uint32_t u[8]; } bf;
        #pragma unroll
        for (int v = 0; v < 4; ++v)                        // ci pairs 0,2,4,6
            bf.u[v] = vA ? *(const uint32_t*)(bptr + offA + 4 * v) : 0u;
        #pragma unroll
        for (int v = 4; v < 8; ++v)
            bf.u[v] = vB ? *(const uint32_t*)(bptr + offB + 4 * (v - 4)) : 0u;

        // ---------------- A fragments (16M x 32K, f16) ----------------
        union { v16h h; uint32_t u[8]; } a0, a1;
        #pragma unroll
        for (int v = 0; v < 8; ++v) {
            int boff = c * 64 + ((v < 4) ? 0 : 32) + (v & 3) * 4;  // immediate
            a0.u[v] = *(const uint32_t*)(aptr0 + boff);
            a1.u[v] = *(const uint32_t*)(aptr1 + boff);
        }

        acc0 = __builtin_amdgcn_wmma_f32_16x16x32_f16(
                   false, a0.h, false, bf.h, (short)0, acc0, false, false);
        acc1 = __builtin_amdgcn_wmma_f32_16x16x32_f16(
                   false, a1.h, false, bf.h, (short)0, acc1, false, false);
    }

    // ---- add bias sums and store (C/D layout: n=lane&15, M=v + (isHi?8:0)) ----
    const int rowbase = isHi ? 8 : 0;
    const int h = h0 + h_off;
    size_t obase = (((((size_t)b * CO) * OO + o) * DD + d) * HH + h) * WW + w;
    #pragma unroll
    for (int v = 0; v < 8; ++v) {
        int co0 = rowbase + v;
        int co1 = 16 + rowbase + v;
        out[obase + (size_t)co0 * COSTRIDE] = acc0[v] + bsum[co0];
        out[obase + (size_t)co1 * COSTRIDE] = acc1[v] + bsum[co1];
    }
}

// ---------------------------------------------------------------------------
extern "C" void kernel_launch(void* const* d_in, const int* in_sizes, int n_in,
                              void* d_out, int out_size, void* d_ws, size_t ws_size,
                              hipStream_t stream) {
    (void)in_sizes; (void)n_in; (void)out_size; (void)ws_size;
    const float* x    = (const float*)d_in[0];
    const float* wsrc = (const float*)d_in[1];
    const float* bias = (const float*)d_in[2];
    float*       out  = (float*)d_out;
    _Float16*    wf16 = (_Float16*)d_ws;    // 32*672*2 = 43008 bytes

    // 1) convert weights to f16 GEMM-ready table
    wcvt_kernel<<<(CO * RPAD + 255) / 256, 256, 0, stream>>>(wsrc, wf16);

    // 2) implicit-GEMM 4D conv
    int grid = BN * OO * DD * (HH / HB);    // 12800
    conv4d_kernel<<<grid, BLOCK, 0, stream>>>(x, wf16, bias, out);
}
